// HIST_18193481466107
// MI455X (gfx1250) — matrix operand order repaired and checked by reading, not compiled, not run on hardware
//
#include <hip/hip_runtime.h>
#include <cstdint>
#include <cstddef>

#define NSTOCK 4096
#define TSTEP 60
#define DFEAT 6
#define NCONCEPT 512
#define MSLOTS 128
#define HID 128

typedef _Float16 half_t;
typedef __attribute__((ext_vector_type(16))) _Float16 v16h;
typedef __attribute__((ext_vector_type(8)))  _Float16 v8h;
typedef __attribute__((ext_vector_type(8)))  float    v8f;

// ---------------------------------------------------------------------------
// WMMA helpers (CDNA5 v_wmma_f32_16x16x32_f16, wave32)
// A fragment (16x32 MxK f16): lane m=lane&15, hf=lane>>4;
//   elems[0..7]  = A[m][k0 + 8*hf + j]        (16B contiguous)
//   elems[8..15] = A[m][k0 + 16 + 8*hf + j]   (16B contiguous)
// B fragment (32x16 KxN f16) loaded from row-major W(N x K) (x @ W^T form):
//   lane n=lane&15, hf=lane>>4; elems[0..7] = W[n0+n][k0 + 8*hf + j], elems[8..15] = W[n0+n][k0 + 16 + 8*hf + j]
// C/D (16x16 f32): lane n=lane&15; elem r -> row r + 8*(lane>>4)
// ---------------------------------------------------------------------------
__device__ __forceinline__ v16h ld_a(const half_t* __restrict__ A, size_t lda, int k0) {
  int lane = threadIdx.x & 31;
  int m = lane & 15, hf = lane >> 4;
  const half_t* p = A + (size_t)m * lda + k0 + 8 * hf;
  v8h lo = *(const v8h*)p;
  v8h hi = *(const v8h*)(p + 16);
  v16h r;
#pragma unroll
  for (int j = 0; j < 8; ++j) { r[j] = lo[j]; r[j + 8] = hi[j]; }
  return r;
}

__device__ __forceinline__ v16h ld_b(const half_t* __restrict__ W, size_t ldw, int k0) {
  int lane = threadIdx.x & 31;
  int n = lane & 15, hf = lane >> 4;
  const half_t* p = W + (size_t)n * ldw + k0 + 8 * hf;
  v8h lo = *(const v8h*)p;
  v8h hi = *(const v8h*)(p + 16);
  v16h r;
#pragma unroll
  for (int j = 0; j < 8; ++j) { r[j] = lo[j]; r[j + 8] = hi[j]; }
  return r;
}

__device__ __forceinline__ v8f wmma_f16(v16h a, v16h b, v8f c) {
  return __builtin_amdgcn_wmma_f32_16x16x32_f16(false, a, false, b, (short)0, c, false, false);
}

// ---------------------------------------------------------------------------
// Generic GEMM: C(MxN) = act(A16(MxK) @ W16(NxK)^T + bias), optional f16 copy.
// M%16==0, N%64==0, K%32==0. One wave computes a 16x64 tile.
// ---------------------------------------------------------------------------
__global__ void k_gemm_bT(const half_t* __restrict__ A, const half_t* __restrict__ W,
                          const float* __restrict__ bias, float* __restrict__ C,
                          half_t* __restrict__ C16, int M, int N, int K, int act)
{
  int wid = blockIdx.x * (blockDim.x >> 5) + (threadIdx.x >> 5);
  int nt64 = N >> 6;
  int mt = wid / nt64;
  int nt = wid % nt64;
  if (mt * 16 >= M) return;
  const half_t* Ab = A + (size_t)mt * 16 * K;
  v8f acc[4];
#pragma unroll
  for (int j = 0; j < 4; ++j)
#pragma unroll
    for (int e = 0; e < 8; ++e) acc[j][e] = 0.f;
  for (int k0 = 0; k0 < K; k0 += 32) {
    v16h a = ld_a(Ab, (size_t)K, k0);
#pragma unroll
    for (int j = 0; j < 4; ++j) {
      v16h b = ld_b(W + (size_t)(nt * 64 + j * 16) * K, (size_t)K, k0);
      acc[j] = wmma_f16(a, b, acc[j]);
    }
  }
  int lane = threadIdx.x & 31;
  int n = lane & 15, hf = lane >> 4;
#pragma unroll
  for (int j = 0; j < 4; ++j) {
    int col = nt * 64 + j * 16 + n;
    float bv = bias ? bias[col] : 0.f;
#pragma unroll
    for (int r = 0; r < 8; ++r) {
      int row = mt * 16 + r + 8 * hf;
      float v = acc[j][r] + bv;
      if (act == 1) v = v >= 0.f ? v : 0.01f * v;
      C[(size_t)row * N + col] = v;
      if (C16) C16[(size_t)row * N + col] = (half_t)v;
    }
  }
}

// ---------------------------------------------------------------------------
// GRU layer 0: gi from x (K=6, VALU), gh via WMMA. 4 waves/block, 16 rows/wave.
// h carried as A-fragments; per-step update round-trips through LDS.
// ---------------------------------------------------------------------------
__global__ void k_gru0(const float* __restrict__ x,       // (4096, 360)
                       const float* __restrict__ Wih,     // (384, 6)
                       const float* __restrict__ bih,
                       const half_t* __restrict__ Whh16,  // (384, 128) f16
                       const float* __restrict__ bhh,
                       half_t* __restrict__ Hout)         // (4096, 60, 128) f16
{
  __shared__ half_t ldsH[4][16][HID];
  __shared__ float  xs[4][16][8];
  int w    = threadIdx.x >> 5;
  int lane = threadIdx.x & 31;
  int row0 = (blockIdx.x * 4 + w) * 16;
  int m = lane & 15, hf = lane >> 4;

  for (int c = lane; c < 16 * HID; c += 32) ((half_t*)ldsH[w])[c] = (half_t)0.f;
  v16h hA[4];
#pragma unroll
  for (int kt = 0; kt < 4; ++kt)
#pragma unroll
    for (int j = 0; j < 16; ++j) hA[kt][j] = (half_t)0.f;
  __syncthreads();

  for (int t = 0; t < TSTEP; ++t) {
    if (lane < 16) {
#pragma unroll
      for (int d = 0; d < DFEAT; ++d)
        xs[w][lane][d] = x[(size_t)(row0 + lane) * (DFEAT * TSTEP) + d * TSTEP + t];
    }
    __syncthreads();
    for (int jt = 0; jt < 8; ++jt) {
      v8f ar, az, ain, ahn;
#pragma unroll
      for (int e = 0; e < 8; ++e) { ar[e] = 0.f; az[e] = 0.f; ain[e] = 0.f; ahn[e] = 0.f; }
      int c = jt * 16 + (lane & 15);
      float wr[DFEAT], wz[DFEAT], wn[DFEAT];
#pragma unroll
      for (int d = 0; d < DFEAT; ++d) {
        wr[d] = Wih[(size_t)c * DFEAT + d];
        wz[d] = Wih[(size_t)(128 + c) * DFEAT + d];
        wn[d] = Wih[(size_t)(256 + c) * DFEAT + d];
      }
#pragma unroll
      for (int r = 0; r < 8; ++r) {
        int mm = r + 8 * hf;
        float s0 = 0.f, s1 = 0.f, s2 = 0.f;
#pragma unroll
        for (int d = 0; d < DFEAT; ++d) {
          float xv = xs[w][mm][d];
          s0 += wr[d] * xv; s1 += wz[d] * xv; s2 += wn[d] * xv;
        }
        ar[r] = s0; az[r] = s1; ain[r] = s2;
      }
#pragma unroll
      for (int kt = 0; kt < 4; ++kt) {
        int k0 = kt * 32;
        ar  = wmma_f16(hA[kt], ld_b(Whh16 + (size_t)(jt * 16) * HID, HID, k0), ar);
        az  = wmma_f16(hA[kt], ld_b(Whh16 + (size_t)(128 + jt * 16) * HID, HID, k0), az);
        ahn = wmma_f16(hA[kt], ld_b(Whh16 + (size_t)(256 + jt * 16) * HID, HID, k0), ahn);
      }
      float br_ = bih[c] + bhh[c];
      float bz_ = bih[128 + c] + bhh[128 + c];
      float bin_ = bih[256 + c], bhn_ = bhh[256 + c];
#pragma unroll
      for (int r = 0; r < 8; ++r) {
        int mm = r + 8 * hf;
        float rg = 1.f / (1.f + __expf(-(ar[r] + br_)));
        float zg = 1.f / (1.f + __expf(-(az[r] + bz_)));
        float ng = tanhf(ain[r] + bin_ + rg * (ahn[r] + bhn_));
        float hp = (float)ldsH[w][mm][c];
        float hv = (1.f - zg) * ng + zg * hp;
        ldsH[w][mm][c] = (half_t)hv;
        Hout[((size_t)(row0 + mm) * TSTEP + t) * HID + c] = (half_t)hv;
      }
    }
    __syncthreads();
#pragma unroll
    for (int kt = 0; kt < 4; ++kt) {
      const half_t* p = &ldsH[w][m][kt * 32 + 8 * hf];
      v8h lo = *(const v8h*)p;
      v8h hi = *(const v8h*)(p + 16);
#pragma unroll
      for (int j = 0; j < 8; ++j) { hA[kt][j] = lo[j]; hA[kt][j + 8] = hi[j]; }
    }
    __syncthreads();
  }
}

// ---------------------------------------------------------------------------
// GRU layer 1: gi = h1_t @ Wih1^T (WMMA) folded into recurrence; emit last h.
// ---------------------------------------------------------------------------
__global__ void k_gru1(const half_t* __restrict__ Hseq,   // (4096, 60, 128) f16
                       const half_t* __restrict__ Wih16,  // (384, 128) f16
                       const float* __restrict__ bih,
                       const half_t* __restrict__ Whh16,  // (384, 128) f16
                       const float* __restrict__ bhh,
                       float* __restrict__ xhid,
                       half_t* __restrict__ xhid16)
{
  __shared__ half_t ldsH[4][16][HID];
  int w    = threadIdx.x >> 5;
  int lane = threadIdx.x & 31;
  int row0 = (blockIdx.x * 4 + w) * 16;
  int m = lane & 15, hf = lane >> 4;

  for (int c = lane; c < 16 * HID; c += 32) ((half_t*)ldsH[w])[c] = (half_t)0.f;
  v16h hA[4];
#pragma unroll
  for (int kt = 0; kt < 4; ++kt)
#pragma unroll
    for (int j = 0; j < 16; ++j) hA[kt][j] = (half_t)0.f;
  __syncthreads();

  for (int t = 0; t < TSTEP; ++t) {
    v16h aX[4];
    const half_t* Ab = Hseq + ((size_t)row0 * TSTEP + t) * HID;
#pragma unroll
    for (int kt = 0; kt < 4; ++kt) aX[kt] = ld_a(Ab, (size_t)TSTEP * HID, kt * 32);
    for (int jt = 0; jt < 8; ++jt) {
      v8f ar, az, ain, ahn;
#pragma unroll
      for (int e = 0; e < 8; ++e) { ar[e] = 0.f; az[e] = 0.f; ain[e] = 0.f; ahn[e] = 0.f; }
#pragma unroll
      for (int kt = 0; kt < 4; ++kt) {
        int k0 = kt * 32;
        ar  = wmma_f16(aX[kt], ld_b(Wih16 + (size_t)(jt * 16) * HID, HID, k0), ar);
        az  = wmma_f16(aX[kt], ld_b(Wih16 + (size_t)(128 + jt * 16) * HID, HID, k0), az);
        ain = wmma_f16(aX[kt], ld_b(Wih16 + (size_t)(256 + jt * 16) * HID, HID, k0), ain);
        ar  = wmma_f16(hA[kt], ld_b(Whh16 + (size_t)(jt * 16) * HID, HID, k0), ar);
        az  = wmma_f16(hA[kt], ld_b(Whh16 + (size_t)(128 + jt * 16) * HID, HID, k0), az);
        ahn = wmma_f16(hA[kt], ld_b(Whh16 + (size_t)(256 + jt * 16) * HID, HID, k0), ahn);
      }
      int c = jt * 16 + (lane & 15);
      float br_ = bih[c] + bhh[c];
      float bz_ = bih[128 + c] + bhh[128 + c];
      float bin_ = bih[256 + c], bhn_ = bhh[256 + c];
#pragma unroll
      for (int r = 0; r < 8; ++r) {
        int mm = r + 8 * hf;
        float rg = 1.f / (1.f + __expf(-(ar[r] + br_)));
        float zg = 1.f / (1.f + __expf(-(az[r] + bz_)));
        float ng = tanhf(ain[r] + bin_ + rg * (ahn[r] + bhn_));
        float hp = (float)ldsH[w][mm][c];
        float hv = (1.f - zg) * ng + zg * hp;
        ldsH[w][mm][c] = (half_t)hv;
        if (t == TSTEP - 1) {
          xhid[(size_t)(row0 + mm) * HID + c] = hv;
          xhid16[(size_t)(row0 + mm) * HID + c] = (half_t)hv;
        }
      }
    }
    __syncthreads();
#pragma unroll
    for (int kt = 0; kt < 4; ++kt) {
      const half_t* p = &ldsH[w][m][kt * 32 + 8 * hf];
      v8h lo = *(const v8h*)p;
      v8h hi = *(const v8h*)(p + 16);
#pragma unroll
      for (int j = 0; j < 8; ++j) { hA[kt][j] = lo[j]; hA[kt][j + 8] = hi[j]; }
    }
    __syncthreads();
  }
}

// --------------------------- small utility kernels -------------------------
__global__ void k_cast(const float* __restrict__ in, half_t* __restrict__ out, int n) {
  int i = blockIdx.x * blockDim.x + threadIdx.x;
  if (i < n) out[i] = (half_t)in[i];
}

__global__ void k_transpose16(const float* __restrict__ in, half_t* __restrict__ out,
                              int M, int N) { // out(N x M) f16 from in(M x N) f32
  int i = blockIdx.x * blockDim.x + threadIdx.x;
  if (i < M * N) { int r = i / N, c = i % N; out[(size_t)c * M + r] = (half_t)in[i]; }
}

__global__ void k_colsum(const float* __restrict__ cm, const float* __restrict__ mv,
                         float* __restrict__ colsum) {
  __shared__ float red[256];
  int c = blockIdx.x, t = threadIdx.x;
  float s = 0.f;
  for (int i = t; i < NSTOCK; i += 256) s += cm[(size_t)i * NCONCEPT + c] * mv[i];
  red[t] = s; __syncthreads();
  for (int o = 128; o > 0; o >>= 1) { if (t < o) red[t] += red[t + o]; __syncthreads(); }
  if (!t) colsum[c] = red[0];
}

__global__ void k_s2c_hidden(const float* __restrict__ cm, const float* __restrict__ mv,
                             const float* __restrict__ xhid, const float* __restrict__ colsum,
                             float* __restrict__ hid, half_t* __restrict__ hid16) {
  int c = blockIdx.x, f = threadIdx.x;
  float inv = 1.f / (colsum[c] + 1.f);
  float acc = 0.f;
  for (int i = 0; i < NSTOCK; ++i) {
    float wv = cm[(size_t)i * NCONCEPT + c];
    if (wv != 0.f) acc += wv * mv[i] * xhid[(size_t)i * HID + f];
  }
  float v = acc * inv;
  hid[(size_t)c * HID + f] = v;
  hid16[(size_t)c * HID + f] = (half_t)v;
}

__global__ void k_colsoft_hidden(const float* __restrict__ sc, const float* __restrict__ xhid,
                                 float* __restrict__ hid, half_t* __restrict__ hid16) {
  __shared__ float red[128];
  int c = blockIdx.x, t = threadIdx.x;
  float mx = -3.4e38f;
  for (int i = t; i < NSTOCK; i += 128) mx = fmaxf(mx, sc[(size_t)i * NCONCEPT + c]);
  red[t] = mx; __syncthreads();
  for (int o = 64; o > 0; o >>= 1) { if (t < o) red[t] = fmaxf(red[t], red[t + o]); __syncthreads(); }
  mx = red[0]; __syncthreads();
  float s = 0.f;
  for (int i = t; i < NSTOCK; i += 128) s += __expf(sc[(size_t)i * NCONCEPT + c] - mx);
  red[t] = s; __syncthreads();
  for (int o = 64; o > 0; o >>= 1) { if (t < o) red[t] += red[t + o]; __syncthreads(); }
  float inv = 1.f / red[0];
  float acc = 0.f;
  for (int i = 0; i < NSTOCK; ++i)
    acc += __expf(sc[(size_t)i * NCONCEPT + c] - mx) * inv * xhid[(size_t)i * HID + t];
  hid[(size_t)c * HID + t] = acc;
  hid16[(size_t)c * HID + t] = (half_t)acc;
}

__global__ void k_rownorm(const float* __restrict__ A, float* __restrict__ nrm) {
  __shared__ float red[128];
  int i = blockIdx.x, t = threadIdx.x;
  float v = A[(size_t)i * HID + t];
  red[t] = v * v; __syncthreads();
  for (int o = 64; o > 0; o >>= 1) { if (t < o) red[t] += red[t + o]; __syncthreads(); }
  if (!t) nrm[i] = sqrtf(red[0]);
}

// cos-sim (optional) + row softmax -> f16 probabilities. xn/yn == null => plain.
__global__ void k_cos_rowsoftmax(const float* __restrict__ sc, const float* __restrict__ xn,
                                 const float* __restrict__ yn, half_t* __restrict__ out, int N) {
  __shared__ float red[256];
  int i = blockIdx.x, t = threadIdx.x;
  float xni = xn ? xn[i] : 1.f;
  const float* row = sc + (size_t)i * N;
  float mx = -3.4e38f;
  for (int j = t; j < N; j += 256) {
    float d = xni * (yn ? yn[j] : 1.f);
    float v = d > 0.f ? row[j] / d : 0.f;
    mx = fmaxf(mx, v);
  }
  red[t] = mx; __syncthreads();
  for (int o = 128; o > 0; o >>= 1) { if (t < o) red[t] = fmaxf(red[t], red[t + o]); __syncthreads(); }
  mx = red[0]; __syncthreads();
  float s = 0.f;
  for (int j = t; j < N; j += 256) {
    float d = xni * (yn ? yn[j] : 1.f);
    float v = d > 0.f ? row[j] / d : 0.f;
    s += __expf(v - mx);
  }
  red[t] = s; __syncthreads();
  for (int o = 128; o > 0; o >>= 1) { if (t < o) red[t] += red[t + o]; __syncthreads(); }
  float inv = 1.f / red[0];
  for (int j = t; j < N; j += 256) {
    float d = xni * (yn ? yn[j] : 1.f);
    float v = d > 0.f ? row[j] / d : 0.f;
    out[(size_t)i * N + j] = (half_t)(__expf(v - mx) * inv);
  }
}

__global__ void k_mul(const float* a, const float* b, float* o, half_t* o16, int n) {
  int i = blockIdx.x * blockDim.x + threadIdx.x;
  if (i < n) { float v = a[i] * b[i]; o[i] = v; o16[i] = (half_t)v; }
}

__global__ void k_sub2(const float* a, const float* b, float* o, half_t* o16, int n) {
  int i = blockIdx.x * blockDim.x + threadIdx.x;
  if (i < n) { float v = a[i] - b[i]; o[i] = v; o16[i] = (half_t)v; }
}

__global__ void k_sub3(const float* a, const float* b, const float* c, float* o, half_t* o16, int n) {
  int i = blockIdx.x * blockDim.x + threadIdx.x;
  if (i < n) { float v = a[i] - b[i] - c[i]; o[i] = v; o16[i] = (half_t)v; }
}

// ----------------------------- memory upload -------------------------------
__global__ void k_colmax(const float* __restrict__ sc, float* __restrict__ cmax) {
  __shared__ float red[256];
  int c = blockIdx.x, t = threadIdx.x;
  float mx = -3.4e38f;
  for (int i = t; i < NSTOCK; i += 256) mx = fmaxf(mx, sc[(size_t)i * MSLOTS + c]);
  red[t] = mx; __syncthreads();
  for (int o = 128; o > 0; o >>= 1) { if (t < o) red[t] = fmaxf(red[t], red[t + o]); __syncthreads(); }
  if (!t) cmax[c] = red[0];
}

__global__ void k_argmax_w(const float* __restrict__ sc, const float* __restrict__ cmax,
                           int* __restrict__ g, float* __restrict__ wv) {
  __shared__ float rv[128]; __shared__ int ri[128];
  int i = blockIdx.x, t = threadIdx.x;
  rv[t] = sc[(size_t)i * MSLOTS + t]; ri[t] = t; __syncthreads();
  for (int o = 64; o > 0; o >>= 1) {
    if (t < o) {
      if (rv[t + o] > rv[t] || (rv[t + o] == rv[t] && ri[t + o] < ri[t])) {
        rv[t] = rv[t + o]; ri[t] = ri[t + o];
      }
    }
    __syncthreads();
  }
  if (!t) { int gg = ri[0]; g[i] = gg; wv[i] = __expf(sc[(size_t)i * MSLOTS + gg] - cmax[gg]); }
}

// deterministic segment-sum + (qu + K) row-normalize -> mem out
__global__ void k_qu_final(const int* __restrict__ g, const float* __restrict__ wv,
                           const float* __restrict__ Q, const float* __restrict__ K0,
                           float* __restrict__ out) {
  __shared__ float red[128];
  int s = blockIdx.x, f = threadIdx.x;
  float acc = 0.f;
  for (int i = 0; i < NSTOCK; ++i)
    if (g[i] == s) acc += wv[i] * Q[(size_t)i * HID + f];
  float v = acc + K0[(size_t)s * HID + f];
  red[f] = v * v; __syncthreads();
  for (int o = 64; o > 0; o >>= 1) { if (f < o) red[f] += red[f + o]; __syncthreads(); }
  float nrm = fmaxf(sqrtf(red[0]), 1e-12f);
  out[(size_t)s * HID + f] = v / nrm;
}

// ----------------------------- top-3 neighbors -----------------------------
__global__ void k_topk(const float* __restrict__ sim, const float* __restrict__ hn,
                       float* __restrict__ val3, int* __restrict__ idx3) {
  __shared__ float sv[768]; __shared__ int si[768];
  __shared__ float tv[3];  __shared__ int ti[3];
  int i = blockIdx.x, t = threadIdx.x;
  float hni = hn[i];
  float v0 = -3.4e38f, v1 = -3.4e38f, v2 = -3.4e38f; int i0 = -1, i1 = -1, i2 = -1;
  for (int j = t; j < NSTOCK; j += 256) {
    float d = hni * hn[j];
    float v = (j == i) ? 0.f : (d > 0.f ? sim[(size_t)i * NSTOCK + j] / d : 0.f);
    if (v > v0) { v2 = v1; i2 = i1; v1 = v0; i1 = i0; v0 = v; i0 = j; }
    else if (v > v1) { v2 = v1; i2 = i1; v1 = v; i1 = j; }
    else if (v > v2) { v2 = v; i2 = j; }
  }
  sv[3 * t] = v0; si[3 * t] = i0;
  sv[3 * t + 1] = v1; si[3 * t + 1] = i1;
  sv[3 * t + 2] = v2; si[3 * t + 2] = i2;
  __syncthreads();
  if (t == 0) {
    float b0 = -3.4e38f, b1 = b0, b2 = b0; int j0 = -1, j1 = -1, j2 = -1;
    for (int e = 0; e < 768; ++e) {
      float v = sv[e]; int j = si[e];
      if (j < 0) continue;
      if (v > b0 || (v == b0 && j < j0)) { b2 = b1; j2 = j1; b1 = b0; j1 = j0; b0 = v; j0 = j; }
      else if (v > b1 || (v == b1 && j < j1)) { b2 = b1; j2 = j1; b1 = v; j1 = j; }
      else if (v > b2 || (v == b2 && j < j2)) { b2 = v; j2 = j; }
    }
    tv[0] = b0; ti[0] = j0; tv[1] = b1; ti[1] = j1; tv[2] = b2; ti[2] = j2;
  }
  __syncthreads();
  if (t < 3) { val3[i * 3 + t] = tv[t]; idx3[i * 3 + t] = ti[t]; }
}

// hidden2[j] = sum_i sim_m[i,j]*hs[i]  (+ diag term), deterministic gather
__global__ void k_hidden2(const float* __restrict__ val3, const int* __restrict__ idx3,
                          const float* __restrict__ hs, const float* __restrict__ hn,
                          float* __restrict__ hidden2, half_t* __restrict__ h2_16) {
  int j = blockIdx.x, f = threadIdx.x;
  float acc = 0.f, cs = 0.f;
  for (int i = 0; i < NSTOCK; ++i) {
    int b = i * 3;
    int a0 = idx3[b], a1 = idx3[b + 1], a2 = idx3[b + 2];
    float hv = 0.f;
    bool any = (a0 == j) | (a1 == j) | (a2 == j);
    if (any) hv = hs[(size_t)i * HID + f];
    if (a0 == j) { float v = val3[b];     acc += v * hv; cs += v; }
    if (a1 == j) { float v = val3[b + 1]; acc += v * hv; cs += v; }
    if (a2 == j) { float v = val3[b + 2]; acc += v * hv; cs += v; }
  }
  if (cs != 0.f && hn[j] > 0.f) acc += hs[(size_t)j * HID + f]; // diag = 1 (cos self)
  hidden2[(size_t)j * HID + f] = acc;
  h2_16[(size_t)j * HID + f] = (half_t)acc;
}

__global__ void k_pred(const float* __restrict__ a, const float* __restrict__ b,
                       const float* __restrict__ c, const float* __restrict__ Wout,
                       const float* __restrict__ bout, float* __restrict__ pred) {
  int i = blockIdx.x * blockDim.x + threadIdx.x;
  if (i >= NSTOCK) return;
  float s = bout[0];
  for (int f = 0; f < HID; ++f)
    s += (a[(size_t)i * HID + f] + b[(size_t)i * HID + f] + c[(size_t)i * HID + f]) * Wout[f];
  pred[i] = s;
}

// ---------------------------------------------------------------------------
extern "C" void kernel_launch(void* const* d_in, const int* in_sizes, int n_in,
                              void* d_out, int out_size, void* d_ws, size_t ws_size,
                              hipStream_t stream) {
  (void)in_sizes; (void)n_in; (void)out_size; (void)ws_size;
  // input order: x, concept_matrix, market_value, m_item0, m_item1, params..., train
  const float* x    = (const float*)d_in[0];
  const float* cm   = (const float*)d_in[1];
  const float* mv   = (const float*)d_in[2];
  const float* m0   = (const float*)d_in[3];
  const float* m1   = (const float*)d_in[4];
  const float* Wih0 = (const float*)d_in[5];
  const float* Whh0 = (const float*)d_in[6];
  const float* bih0 = (const float*)d_in[7];
  const float* bhh0 = (const float*)d_in[8];
  const float* Wih1 = (const float*)d_in[9];
  const float* Whh1 = (const float*)d_in[10];
  const float* bih1 = (const float*)d_in[11];
  const float* bhh1 = (const float*)d_in[12];
  const float* W_ps = (const float*)d_in[13];  const float* b_ps = (const float*)d_in[14];
  const float* W_hs = (const float*)d_in[15];  const float* b_hs = (const float*)d_in[16];
  const float* W_psf = (const float*)d_in[17]; const float* b_psf = (const float*)d_in[18];
  const float* W_hsf = (const float*)d_in[19]; const float* b_hsf = (const float*)d_in[20];
  const float* W_psb = (const float*)d_in[21]; const float* b_psb = (const float*)d_in[22];
  const float* W_hsb = (const float*)d_in[23]; const float* b_hsb = (const float*)d_in[24];
  const float* W_indi = (const float*)d_in[25]; const float* b_indi = (const float*)d_in[26];
  const float* W_out = (const float*)d_in[27]; const float* b_out = (const float*)d_in[28];

  float* outp = (float*)d_out;          // pred (4096)
  float* out_mem0 = outp + NSTOCK;      // (128,128)
  float* out_mem1 = out_mem0 + MSLOTS * HID;

  // -------- workspace layout --------
  char* wsp = (char*)d_ws;
  auto bump = [&](size_t bytes) -> char* {
    char* r = wsp; wsp += (bytes + 255) & ~(size_t)255; return r;
  };
  half_t* H1_16   = (half_t*)bump((size_t)NSTOCK * TSTEP * HID * 2);
  half_t* Whh0_16 = (half_t*)bump(384 * 128 * 2);
  half_t* Wih1_16 = (half_t*)bump(384 * 128 * 2);
  half_t* Whh1_16 = (half_t*)bump(384 * 128 * 2);
  half_t* Wps16 = (half_t*)bump(HID * HID * 2);
  half_t* Whs16 = (half_t*)bump(HID * HID * 2);
  half_t* Wpsf16 = (half_t*)bump(HID * HID * 2);
  half_t* Whsf16 = (half_t*)bump(HID * HID * 2);
  half_t* Wpsb16 = (half_t*)bump(HID * HID * 2);
  half_t* Whsb16 = (half_t*)bump(HID * HID * 2);
  half_t* Windi16 = (half_t*)bump(HID * HID * 2);
  half_t* m0_16 = (half_t*)bump(MSLOTS * HID * 2);
  half_t* m0T16 = (half_t*)bump(MSLOTS * HID * 2);
  half_t* m1_16 = (half_t*)bump(MSLOTS * HID * 2);
  half_t* m1T16 = (half_t*)bump(MSLOTS * HID * 2);
  float*  xhid = (float*)bump((size_t)NSTOCK * HID * 4);
  half_t* xhid16 = (half_t*)bump((size_t)NSTOCK * HID * 2);
  float*  colsum = (float*)bump(NCONCEPT * 4);
  float*  hidA = (float*)bump(NCONCEPT * HID * 4);
  half_t* hidA16 = (half_t*)bump(NCONCEPT * HID * 2);
  float*  hidB = (float*)bump(NCONCEPT * HID * 4);
  half_t* hidB16 = (half_t*)bump(NCONCEPT * HID * 2);
  half_t* hidBT16 = (half_t*)bump(NCONCEPT * HID * 2);
  float*  sc512 = (float*)bump((size_t)NSTOCK * NCONCEPT * 4);
  float*  xn = (float*)bump(NSTOCK * 4);
  float*  yn512 = (float*)bump(NCONCEPT * 4);
  half_t* c2s16 = (half_t*)bump((size_t)NSTOCK * NCONCEPT * 2);
  float*  p0 = (float*)bump((size_t)NSTOCK * HID * 4);
  half_t* p016 = (half_t*)bump((size_t)NSTOCK * HID * 2);
  float*  p = (float*)bump((size_t)NSTOCK * HID * 4);
  half_t* p16 = (half_t*)bump((size_t)NSTOCK * HID * 2);
  float*  scm = (float*)bump((size_t)NSTOCK * MSLOTS * 4);
  half_t* ssm16 = (half_t*)bump((size_t)NSTOCK * MSLOTS * 2);
  float*  readv = (float*)bump((size_t)NSTOCK * HID * 4);
  float*  pR = (float*)bump((size_t)NSTOCK * HID * 4);
  half_t* pR16 = (half_t*)bump((size_t)NSTOCK * HID * 2);
  float*  cmax = (float*)bump(MSLOTS * 4);
  int*    gidx = (int*)bump(NSTOCK * 4);
  float*  wvv = (float*)bump(NSTOCK * 4);
  float*  p_back = (float*)bump((size_t)NSTOCK * HID * 4);
  float*  out_ps = (float*)bump((size_t)NSTOCK * HID * 4);
  float*  hs = (float*)bump((size_t)NSTOCK * HID * 4);
  half_t* hs16 = (half_t*)bump((size_t)NSTOCK * HID * 2);
  float*  hn = (float*)bump(NSTOCK * 4);
  float*  simbuf = (float*)bump((size_t)NSTOCK * NSTOCK * 4);
  float*  val3 = (float*)bump(NSTOCK * 3 * 4);
  int*    idx3 = (int*)bump(NSTOCK * 3 * 4);
  float*  hidden2 = (float*)bump((size_t)NSTOCK * HID * 4);
  half_t* h2_16 = (half_t*)bump((size_t)NSTOCK * HID * 2);
  half_t* h2T16 = (half_t*)bump((size_t)NSTOCK * HID * 2);
  float*  h2n = (float*)bump(NSTOCK * 4);
  half_t* c2s2_16 = (half_t*)bump((size_t)NSTOCK * NSTOCK * 2);
  float*  h0 = (float*)bump((size_t)NSTOCK * HID * 4);
  half_t* h016 = (half_t*)bump((size_t)NSTOCK * HID * 2);
  float*  hbuf = (float*)bump((size_t)NSTOCK * HID * 4);
  half_t* h16 = (half_t*)bump((size_t)NSTOCK * HID * 2);
  float*  hR = (float*)bump((size_t)NSTOCK * HID * 4);
  half_t* hR16 = (half_t*)bump((size_t)NSTOCK * HID * 2);
  float*  h_back = (float*)bump((size_t)NSTOCK * HID * 4);
  float*  out_hs = (float*)bump((size_t)NSTOCK * HID * 4);
  float*  indi = (float*)bump((size_t)NSTOCK * HID * 4);
  half_t* indi16 = (half_t*)bump((size_t)NSTOCK * HID * 2);
  float*  out_indi = (float*)bump((size_t)NSTOCK * HID * 4);

  auto cast = [&](const float* src, half_t* dst, int n) {
    k_cast<<<(n + 255) / 256, 256, 0, stream>>>(src, dst, n);
  };
  auto gemm = [&](const half_t* A, const half_t* W, const float* bias, float* C,
                  half_t* C16, int M, int N, int K, int act) {
    int waves = (M / 16) * (N / 64);
    k_gemm_bT<<<(waves + 7) / 8, 256, 0, stream>>>(A, W, bias, C, C16, M, N, K, act);
  };

  // -------- weight precision staging --------
  cast(Whh0, Whh0_16, 384 * 128);
  cast(Wih1, Wih1_16, 384 * 128);
  cast(Whh1, Whh1_16, 384 * 128);
  cast(W_ps, Wps16, HID * HID);   cast(W_hs, Whs16, HID * HID);
  cast(W_psf, Wpsf16, HID * HID); cast(W_hsf, Whsf16, HID * HID);
  cast(W_psb, Wpsb16, HID * HID); cast(W_hsb, Whsb16, HID * HID);
  cast(W_indi, Windi16, HID * HID);
  cast(m0, m0_16, MSLOTS * HID);  cast(m1, m1_16, MSLOTS * HID);
  k_transpose16<<<(MSLOTS * HID + 255) / 256, 256, 0, stream>>>(m0, m0T16, MSLOTS, HID);
  k_transpose16<<<(MSLOTS * HID + 255) / 256, 256, 0, stream>>>(m1, m1T16, MSLOTS, HID);

  // -------- GRU (WMMA recurrence) --------
  k_gru0<<<NSTOCK / 64, 128, 0, stream>>>(x, Wih0, bih0, Whh0_16, bhh0, H1_16);
  k_gru1<<<NSTOCK / 64, 128, 0, stream>>>(H1_16, Wih1_16, bih1, Whh1_16, bhh1, xhid, xhid16);

  // -------- concept aggregation --------
  k_colsum<<<NCONCEPT, 256, 0, stream>>>(cm, mv, colsum);
  k_s2c_hidden<<<NCONCEPT, HID, 0, stream>>>(cm, mv, xhid, colsum, hidA, hidA16);
  gemm(xhid16, hidA16, nullptr, sc512, nullptr, NSTOCK, NCONCEPT, HID, 0);
  k_colsoft_hidden<<<NCONCEPT, HID, 0, stream>>>(sc512, xhid, hidB, hidB16);
  gemm(xhid16, hidB16, nullptr, sc512, nullptr, NSTOCK, NCONCEPT, HID, 0);
  k_rownorm<<<NSTOCK, HID, 0, stream>>>(xhid, xn);
  k_rownorm<<<NCONCEPT, HID, 0, stream>>>(hidB, yn512);
  k_cos_rowsoftmax<<<NSTOCK, 256, 0, stream>>>(sc512, xn, yn512, c2s16, NCONCEPT);
  k_transpose16<<<(NCONCEPT * HID + 255) / 256, 256, 0, stream>>>(hidB, hidBT16, NCONCEPT, HID);
  gemm(c2s16, hidBT16, nullptr, p0, p016, NSTOCK, HID, NCONCEPT, 0);
  gemm(p016, Wps16, b_ps, p, p16, NSTOCK, HID, HID, 0);

  // -------- memory read + upload (item0) --------
  gemm(p16, m0_16, nullptr, scm, nullptr, NSTOCK, MSLOTS, HID, 0);
  k_cos_rowsoftmax<<<NSTOCK, 256, 0, stream>>>(scm, nullptr, nullptr, ssm16, MSLOTS);
  gemm(ssm16, m0T16, nullptr, readv, nullptr, NSTOCK, HID, MSLOTS, 0);
  k_mul<<<(NSTOCK * HID + 255) / 256, 256, 0, stream>>>(p, readv, pR, pR16, NSTOCK * HID);
  gemm(pR16, m0_16, nullptr, scm, nullptr, NSTOCK, MSLOTS, HID, 0);
  k_colmax<<<MSLOTS, 256, 0, stream>>>(scm, cmax);
  k_argmax_w<<<NSTOCK, MSLOTS, 0, stream>>>(scm, cmax, gidx, wvv);
  k_qu_final<<<MSLOTS, HID, 0, stream>>>(gidx, wvv, pR, m0, out_mem0);

  gemm(pR16, Wpsb16, b_psb, p_back, nullptr, NSTOCK, HID, HID, 0);
  gemm(pR16, Wpsf16, b_psf, out_ps, nullptr, NSTOCK, HID, HID, 1);
  k_sub2<<<(NSTOCK * HID + 255) / 256, 256, 0, stream>>>(xhid, p_back, hs, hs16, NSTOCK * HID);

  // -------- neighbor graph (4096x4096 WMMA) --------
  k_rownorm<<<NSTOCK, HID, 0, stream>>>(hs, hn);
  gemm(hs16, hs16, nullptr, simbuf, nullptr, NSTOCK, NSTOCK, HID, 0);
  k_topk<<<NSTOCK, 256, 0, stream>>>(simbuf, hn, val3, idx3);
  k_hidden2<<<NSTOCK, HID, 0, stream>>>(val3, idx3, hs, hn, hidden2, h2_16);
  k_rownorm<<<NSTOCK, HID, 0, stream>>>(hidden2, h2n);
  gemm(hs16, h2_16, nullptr, simbuf, nullptr, NSTOCK, NSTOCK, HID, 0);
  k_cos_rowsoftmax<<<NSTOCK, 256, 0, stream>>>(simbuf, hn, h2n, c2s2_16, NSTOCK);
  k_transpose16<<<(NSTOCK * HID + 255) / 256, 256, 0, stream>>>(hidden2, h2T16, NSTOCK, HID);
  gemm(c2s2_16, h2T16, nullptr, h0, h016, NSTOCK, HID, NSTOCK, 0);
  gemm(h016, Whs16, b_hs, hbuf, h16, NSTOCK, HID, HID, 0);

  // -------- memory read + upload (item1) --------
  gemm(h16, m1_16, nullptr, scm, nullptr, NSTOCK, MSLOTS, HID, 0);
  k_cos_rowsoftmax<<<NSTOCK, 256, 0, stream>>>(scm, nullptr, nullptr, ssm16, MSLOTS);
  gemm(ssm16, m1T16, nullptr, readv, nullptr, NSTOCK, HID, MSLOTS, 0);
  k_mul<<<(NSTOCK * HID + 255) / 256, 256, 0, stream>>>(hbuf, readv, hR, hR16, NSTOCK * HID);
  gemm(hR16, m1_16, nullptr, scm, nullptr, NSTOCK, MSLOTS, HID, 0);
  k_colmax<<<MSLOTS, 256, 0, stream>>>(scm, cmax);
  k_argmax_w<<<NSTOCK, MSLOTS, 0, stream>>>(scm, cmax, gidx, wvv);
  k_qu_final<<<MSLOTS, HID, 0, stream>>>(gidx, wvv, hR, m1, out_mem1);

  // -------- heads --------
  gemm(hR16, Whsb16, b_hsb, h_back, nullptr, NSTOCK, HID, HID, 0);
  gemm(hR16, Whsf16, b_hsf, out_hs, nullptr, NSTOCK, HID, HID, 1);
  k_sub3<<<(NSTOCK * HID + 255) / 256, 256, 0, stream>>>(xhid, p_back, h_back, indi, indi16,
                                                         NSTOCK * HID);
  gemm(indi16, Windi16, b_indi, out_indi, nullptr, NSTOCK, HID, HID, 1);
  k_pred<<<NSTOCK / 128, 128, 0, stream>>>(out_ps, out_hs, out_indi, W_out, b_out, outp);
}
